// MultilayerDeformableC3D_53300544143736
// MI455X (gfx1250) — matrix operand order repaired and checked by reading, not compiled, not run on hardware
//
#include <hip/hip_runtime.h>
#include <hip/hip_bf16.h>

typedef __bf16 bf16;
typedef __attribute__((ext_vector_type(16))) __bf16 v16bf;
typedef __attribute__((ext_vector_type(8)))  float  v8f;

#define TD   8
#define HD   24
#define WD   24
#define NN   4608          // TD*HD*WD
#define CIN  256
#define KTOT 6912          // 27*CIN
#define BM   128
#define BN   128
#define BK   32
#define LDP  40            // LDS row pitch (bf16) -> 80B = 20-bank stride, conflict-free

// ---- WMMA fragment loads from LDS (layouts per CDNA5 ISA 7.12.2) ----------
// A 16x32 bf16: lane half=L/16, row m=L%16; elems 0..7 -> K=half*8+j, 8..15 -> K=16+half*8+j
__device__ __forceinline__ v16bf frag_a(const bf16* rowbase, int half) {
  union { float4 f4[2]; v16bf v; } u;
  u.f4[0] = *(const float4*)(rowbase + half * 8);
  u.f4[1] = *(const float4*)(rowbase + 16 + half * 8);
  return u.v;
}
// B 32x16 bf16: lane half=L/16, col n=L%16; elems 0..7 -> K=half*8+j, 8..15 -> K=16+half*8+j
__device__ __forceinline__ v16bf frag_b(const bf16* rowbase, int half) {
  union { float4 f4[2]; v16bf v; } u;
  u.f4[0] = *(const float4*)(rowbase + half * 8);
  u.f4[1] = *(const float4*)(rowbase + 16 + half * 8);
  return u.v;
}

// ---------------------------------------------------------------------------
// Implicit-GEMM conv / plain GEMM with bf16 WMMA, f32 accumulate.
//   A    : [M][KTOT] bf16, K ordered kk*CIN+ci (kk-major)
//   B    : IM2COL: activation [NN][CIN] bf16 (transposed layout)
//          else  : acc matrix [NN][KTOT] bf16
//   EPI  : 0 = bf16 transposed out [NN][M], bias+ReLU   (conv layers)
//          1 = f32  transposed out [NN][M], bias        (offset conv)
//          2 = f32  m-major    out [M][NN], bias        (final output)
// ---------------------------------------------------------------------------
template <bool IM2COL, int EPI>
__global__ __launch_bounds__(256)
void gemm_wmma(const bf16* __restrict__ A, const bf16* __restrict__ B,
               const float* __restrict__ bias, void* __restrict__ Out, int M)
{
  __shared__ bf16 As[2][BM * LDP];
  __shared__ bf16 Bs[2][BN * LDP];

  const int tid = threadIdx.x;
  const int n0  = blockIdx.x * BN;
  const int m0  = blockIdx.y * BM;

  // staging-load mapping: 256 threads cover 128 rows x 2 k-halves (16 bf16 each)
  const int lrow  = tid & 127;
  const int lhalf = tid >> 7;

  // wave mapping: 8 waves in 2(M) x 4(N); wave tile 64x32 = 4x2 fragments
  const int wid   = tid >> 5;
  const int lane  = tid & 31;
  const int wm    = (wid & 1) * 64;
  const int wn    = (wid >> 1) * 32;
  const int fhalf = lane >> 4;
  const int fr    = lane & 15;

  v8f acc[4][2];
#pragma unroll
  for (int i = 0; i < 4; ++i)
#pragma unroll
    for (int j = 0; j < 2; ++j) acc[i][j] = (v8f){0,0,0,0,0,0,0,0};

  // spatial decode of this thread's B-row (implicit im2col)
  int bt = 0, bh = 0, bw = 0;
  if (IM2COL) {
    const int n = n0 + lrow;
    bt = n / (HD * WD);
    const int r = n - bt * (HD * WD);
    bh = r / WD;
    bw = r - bh * WD;
  }

  const int steps = KTOT / BK;  // 216
  float4 a0, a1, b0, b1;

  auto load_tile = [&](int kb) {
    const bf16* ap = A + (size_t)(m0 + lrow) * KTOT + kb * BK + lhalf * 16;
    a0 = ((const float4*)ap)[0];
    a1 = ((const float4*)ap)[1];
    if (IM2COL) {
      const int kk  = kb >> 3;                       // 8 k-steps per tap
      const int ci0 = ((kb & 7) * 32) + lhalf * 16;  // channel slice
      const int tt = bt + (kk / 9) - 1;
      const int hh = bh + ((kk / 3) % 3) - 1;
      const int ww = bw + (kk % 3) - 1;
      if ((unsigned)tt < TD && (unsigned)hh < HD && (unsigned)ww < WD) {
        const bf16* bp = B + (size_t)((tt * HD + hh) * WD + ww) * CIN + ci0;
        b0 = ((const float4*)bp)[0];
        b1 = ((const float4*)bp)[1];
      } else {
        b0 = make_float4(0.f, 0.f, 0.f, 0.f);
        b1 = make_float4(0.f, 0.f, 0.f, 0.f);
      }
    } else {
      const bf16* bp = B + (size_t)(n0 + lrow) * KTOT + kb * BK + lhalf * 16;
      b0 = ((const float4*)bp)[0];
      b1 = ((const float4*)bp)[1];
    }
  };
  auto store_tile = [&](int buf) {
    *(float4*)(&As[buf][lrow * LDP + lhalf * 16 + 0]) = a0;
    *(float4*)(&As[buf][lrow * LDP + lhalf * 16 + 8]) = a1;
    *(float4*)(&Bs[buf][lrow * LDP + lhalf * 16 + 0]) = b0;
    *(float4*)(&Bs[buf][lrow * LDP + lhalf * 16 + 8]) = b1;
  };

  load_tile(0);
  store_tile(0);
  __syncthreads();

  for (int kb = 0; kb < steps; ++kb) {
    const int cur = kb & 1;
    if (kb + 1 < steps) load_tile(kb + 1);
    if (kb + 2 < steps) {
      __builtin_prefetch(A + (size_t)(m0 + lrow) * KTOT + (kb + 2) * BK, 0, 1);
      if (!IM2COL)
        __builtin_prefetch(B + (size_t)(n0 + lrow) * KTOT + (kb + 2) * BK, 0, 1);
    }

    v16bf af[4], bfr[2];
#pragma unroll
    for (int i = 0; i < 4; ++i)
      af[i] = frag_a(&As[cur][(wm + i * 16 + fr) * LDP], fhalf);
#pragma unroll
    for (int j = 0; j < 2; ++j)
      bfr[j] = frag_b(&Bs[cur][(wn + j * 16 + fr) * LDP], fhalf);

#pragma unroll
    for (int i = 0; i < 4; ++i)
#pragma unroll
      for (int j = 0; j < 2; ++j)
        acc[i][j] = __builtin_amdgcn_wmma_f32_16x16x32_bf16(
            false, af[i], false, bfr[j], (short)0, acc[i][j], false, false);

    if (kb + 1 < steps) {
      store_tile(cur ^ 1);
      __syncthreads();
    }
  }

  // ---- epilogue: lane holds N = fr, M = base + 0..7 (consecutive) ----
#pragma unroll
  for (int i = 0; i < 4; ++i) {
    const int mg = m0 + wm + i * 16 + fhalf * 8;
    float bv[8];
    *(float4*)&bv[0] = ((const float4*)(bias + mg))[0];
    *(float4*)&bv[4] = ((const float4*)(bias + mg))[1];
#pragma unroll
    for (int j = 0; j < 2; ++j) {
      const int ng = n0 + wn + j * 16 + fr;
      v8f c = acc[i][j];
      if (EPI == 0) {
        union { bf16 h[8]; float4 f; } o;
#pragma unroll
        for (int r = 0; r < 8; ++r)
          o.h[r] = (bf16)fmaxf(c[r] + bv[r], 0.f);
        *(float4*)((bf16*)Out + (size_t)ng * M + mg) = o.f;
      } else if (EPI == 1) {
        float o8[8];
#pragma unroll
        for (int r = 0; r < 8; ++r) o8[r] = c[r] + bv[r];
        float* op = (float*)Out + (size_t)ng * M + mg;
        *(float4*)(op + 0) = *(float4*)&o8[0];
        *(float4*)(op + 4) = *(float4*)&o8[4];
      } else {
#pragma unroll
        for (int r = 0; r < 8; ++r)
          ((float*)Out)[(size_t)(mg + r) * NN + ng] = c[r] + bv[r];
      }
    }
  }
}

// ---- trilinear deformable sampling: acc[n][kk*256+ci] --------------------
__global__ __launch_bounds__(256)
void deform_sample(const float* __restrict__ xt,   // [NN][CIN] f32
                   const float* __restrict__ offT, // [NN][128] f32 (81 valid)
                   bf16* __restrict__ accT)        // [NN][KTOT] bf16
{
  const int n  = blockIdx.x;
  const int ci = threadIdx.x;
  const int t = n / (HD * WD);
  const int r = n - t * (HD * WD);
  const int h = r / WD;
  const int w = r - h * WD;
  const float* offn = offT + (size_t)n * 128;
  bf16* out = accT + (size_t)n * KTOT + ci;

  for (int kk = 0; kk < 27; ++kk) {
    const float pt = (float)(t + (kk / 9) - 1)       + offn[kk * 3 + 0];
    const float ph = (float)(h + ((kk / 3) % 3) - 1) + offn[kk * 3 + 1];
    const float pw = (float)(w + (kk % 3) - 1)       + offn[kk * 3 + 2];
    const float ft = floorf(pt), fh = floorf(ph), fw = floorf(pw);
    const float at = pt - ft, ah = ph - fh, aw = pw - fw;
    const int it = (int)ft, ih = (int)fh, iw = (int)fw;
    float accv = 0.f;
#pragma unroll
    for (int dt = 0; dt < 2; ++dt) {
      const int ti = it + dt;
      const float wt = dt ? at : 1.f - at;
      const bool vt = (unsigned)ti < (unsigned)TD;
      const int tc = min(max(ti, 0), TD - 1);
#pragma unroll
      for (int dh = 0; dh < 2; ++dh) {
        const int hi = ih + dh;
        const float wh = dh ? ah : 1.f - ah;
        const bool vh = (unsigned)hi < (unsigned)HD;
        const int hc = min(max(hi, 0), HD - 1);
#pragma unroll
        for (int dw = 0; dw < 2; ++dw) {
          const int wi = iw + dw;
          const float wwt = dw ? aw : 1.f - aw;
          const bool vw = (unsigned)wi < (unsigned)WD;
          const int wc = min(max(wi, 0), WD - 1);
          const float wgt = (vt && vh && vw) ? wt * wh * wwt : 0.f;
          accv += wgt * xt[(size_t)((tc * HD + hc) * WD + wc) * CIN + ci];
        }
      }
    }
    out[kk * CIN] = (bf16)accv;
  }
}

// ---- setup kernels --------------------------------------------------------
__global__ void transpose_cast(const float* __restrict__ x,
                               float* __restrict__ xt32, bf16* __restrict__ xtbf)
{
  const int idx = blockIdx.x * 256 + threadIdx.x;
  if (idx >= CIN * NN) return;
  const int c = idx / NN, n = idx - c * NN;
  const float v = x[idx];
  xt32[(size_t)n * CIN + c] = v;
  xtbf[(size_t)n * CIN + c] = (bf16)v;
}

// w: [Msrc][CIN][27] f32 -> wr: [Mpad][KTOT] bf16 with k = kk*CIN + ci
__global__ void reorder_w(const float* __restrict__ w, bf16* __restrict__ wr, int Msrc)
{
  const size_t idx = (size_t)blockIdx.x * 256 + threadIdx.x;
  const int m  = (int)(idx / KTOT);
  const int k  = (int)(idx - (size_t)m * KTOT);
  const int kk = k >> 8;       // /CIN
  const int ci = k & 255;
  const float v = (m < Msrc) ? w[((size_t)m * CIN + ci) * 27 + kk] : 0.f;
  wr[idx] = (bf16)v;
}

__global__ void pad_bias(const float* __restrict__ b, float* __restrict__ out,
                         int Msrc, int Mpad)
{
  const int i = threadIdx.x;
  if (i < Mpad) out[i] = (i < Msrc) ? b[i] : 0.f;
}

// ---------------------------------------------------------------------------
extern "C" void kernel_launch(void* const* d_in, const int* in_sizes, int n_in,
                              void* d_out, int out_size, void* d_ws, size_t ws_size,
                              hipStream_t stream) {
  const float* x    = (const float*)d_in[0];
  const float* l1w  = (const float*)d_in[1];
  const float* l1b  = (const float*)d_in[2];
  const float* l2w  = (const float*)d_in[3];
  const float* l2b  = (const float*)d_in[4];
  const float* l3w  = (const float*)d_in[5];
  const float* l3b  = (const float*)d_in[6];
  const float* defw = (const float*)d_in[7];
  const float* defb = (const float*)d_in[8];
  const float* c3w  = (const float*)d_in[9];
  const float* c3b  = (const float*)d_in[10];

  char* ws = (char*)d_ws;
  auto take = [&](size_t bytes) -> char* {
    char* p = ws;
    ws += (bytes + 255) & ~(size_t)255;
    return p;
  };
  float* xt32 = (float*)take((size_t)NN * CIN * 4);
  bf16*  xtbf = (bf16*) take((size_t)NN * CIN * 2);
  bf16*  actA = (bf16*) take((size_t)NN * CIN * 2);
  bf16*  actB = (bf16*) take((size_t)NN * CIN * 2);
  float* offT = (float*)take((size_t)NN * 128 * 4);
  bf16*  accT = (bf16*) take((size_t)NN * KTOT * 2);
  bf16*  wr1  = (bf16*) take((size_t)256 * KTOT * 2);
  bf16*  wr2  = (bf16*) take((size_t)256 * KTOT * 2);
  bf16*  wr3  = (bf16*) take((size_t)256 * KTOT * 2);
  bf16*  wrD  = (bf16*) take((size_t)128 * KTOT * 2);
  bf16*  wrC  = (bf16*) take((size_t)256 * KTOT * 2);
  float* bD   = (float*)take(128 * 4);

  transpose_cast<<<(CIN * NN + 255) / 256, 256, 0, stream>>>(x, xt32, xtbf);
  reorder_w<<<(256 * KTOT) / 256, 256, 0, stream>>>(l1w, wr1, 256);
  reorder_w<<<(256 * KTOT) / 256, 256, 0, stream>>>(l2w, wr2, 256);
  reorder_w<<<(256 * KTOT) / 256, 256, 0, stream>>>(l3w, wr3, 256);
  reorder_w<<<(128 * KTOT) / 256, 256, 0, stream>>>(defw, wrD, 81);
  reorder_w<<<(256 * KTOT) / 256, 256, 0, stream>>>(c3w, wrC, 256);
  pad_bias<<<1, 128, 0, stream>>>(defb, bD, 81, 128);

  dim3 g256(NN / BN, 2), g128(NN / BN, 1);
  gemm_wmma<true, 0><<<g256, 256, 0, stream>>>(wr1, xtbf, l1b, actA, 256);
  gemm_wmma<true, 0><<<g256, 256, 0, stream>>>(wr2, actA, l2b, actB, 256);
  gemm_wmma<true, 0><<<g256, 256, 0, stream>>>(wr3, actB, l3b, actA, 256);
  gemm_wmma<true, 1><<<g128, 256, 0, stream>>>(wrD, actA, bD, offT, 128);
  deform_sample<<<NN, 256, 0, stream>>>(xt32, offT, accT);
  gemm_wmma<false, 2><<<g256, 256, 0, stream>>>(wrC, accT, c3b, (float*)d_out, 256);
}
